// GNN_geoGCN_81758997447248
// MI455X (gfx1250) — matrix operand order repaired and checked by reading, not compiled, not run on hardware
//
#include <hip/hip_runtime.h>
#include <hip/hip_bf16.h>

// ---------------- problem constants ----------------
#define NN      10000          // nodes
#define NE      80000          // edges
#define FIN     128
#define H1      2048
#define H2      2048
#define H3      1024
#define MPAD    10112          // 79 * 128, node dim padded to GEMM tile

// Use CDNA5 async global->LDS staging in the GEMM (ASYNCcnt-tracked DMA,
// no VGPR round-trip). Flip to 0 to revert to the sync ds_store path.
#define USE_ASYNC_STAGING 1

// ---------------- vector types ----------------
typedef __attribute__((ext_vector_type(16))) __bf16 bf16x16;
typedef __attribute__((ext_vector_type(8)))  __bf16 bf16x8;
typedef __attribute__((ext_vector_type(8)))  float  f32x8;

__device__ __forceinline__ unsigned short f32_to_bf16_rne(float f) {
    union { float f; unsigned u; } c; c.f = f;
    unsigned u = c.u;
    unsigned r = u + 0x7FFFu + ((u >> 16) & 1u);   // round-to-nearest-even
    return (unsigned short)(r >> 16);
}

// ---------------- utility kernels ----------------
__global__ void zero_f32(float* __restrict__ p, size_t n) {
    size_t i = (size_t)blockIdx.x * blockDim.x + threadIdx.x;
    size_t stride = (size_t)gridDim.x * blockDim.x;
    for (; i < n; i += stride) p[i] = 0.0f;
}

__global__ void cvt_f32_to_bf16(const float* __restrict__ in,
                                unsigned short* __restrict__ out, size_t n) {
    size_t i = (size_t)blockIdx.x * blockDim.x + threadIdx.x;
    size_t stride = (size_t)gridDim.x * blockDim.x;
    for (; i < n; i += stride) out[i] = f32_to_bf16_rne(in[i]);
}

// W [K,N] fp32 row-major -> Wt [N,K] bf16 row-major  (B pre-transposed for GEMM)
__global__ void cvt_transpose_w(const float* __restrict__ W,
                                unsigned short* __restrict__ Wt,
                                int K, int N) {
    size_t total = (size_t)K * N;
    size_t i = (size_t)blockIdx.x * blockDim.x + threadIdx.x;
    size_t stride = (size_t)gridDim.x * blockDim.x;
    for (; i < total; i += stride) {
        int k = (int)(i % K);
        int n = (int)(i / K);
        Wt[(size_t)n * K + k] = f32_to_bf16_rne(W[(size_t)k * N + n]);
    }
}

// ---------------- edge phase: gate + gather + scatter-add ----------------
// one block per edge; per channel: s = relu(rel . W_in[:,c] + b_in[c]);
// agg[dst][c] += s * x[src][c]
__global__ void edge_scatter(const float* __restrict__ x, int ldx,
                             const float* __restrict__ pos,
                             const int* __restrict__ src,
                             const int* __restrict__ dst,
                             const float* __restrict__ W_in,   // [2,C]
                             const float* __restrict__ b_in,   // [C]
                             float* __restrict__ agg, int ldagg, int C) {
    int e = blockIdx.x;
    int s = src[e];
    int d = dst[e];
    float rx = pos[2 * s]     - pos[2 * d];
    float ry = pos[2 * s + 1] - pos[2 * d + 1];
    const float* w0 = W_in;
    const float* w1 = W_in + C;
    const float* xs = x + (size_t)s * ldx;
    float* ad = agg + (size_t)d * ldagg;
    for (int c = threadIdx.x; c < C; c += blockDim.x) {
        float g = fmaf(rx, w0[c], fmaf(ry, w1[c], b_in[c]));
        g = g > 0.0f ? g : 0.0f;
        atomicAdd(&ad[c], g * xs[c]);
    }
}

// ---------------- bf16 WMMA GEMM: C = relu(A @ B^T + bias) ----------------
// A  : [M,K] bf16 row-major (M = MPAD, multiple of 128; K multiple of 32)
// Bt : [N,K] bf16 row-major (weights pre-transposed; N multiple of 128)
// out: [M,N] fp32
#define BM 128
#define BN 128
#define BK 32
#define LDT 40   // padded LDS row stride (elements): 80B, keeps 16B alignment

#if USE_ASYNC_STAGING
// CDNA5 async copy: 16B global -> LDS, per-lane addresses, tracked by ASYNCcnt.
__device__ __forceinline__ void async_copy_b128(const unsigned short* gp,
                                                unsigned short* lp) {
    unsigned lds_addr = (unsigned)(unsigned long long)lp;  // LDS aperture: low 32b
    unsigned long long gaddr = (unsigned long long)gp;
    asm volatile("global_load_async_to_lds_b128 %0, %1, off"
                 :: "v"(lds_addr), "v"(gaddr) : "memory");
}
#endif

__global__ void __launch_bounds__(256)
gemm_bf16_relu(const unsigned short* __restrict__ A,
               const unsigned short* __restrict__ Bt,
               const float* __restrict__ bias,
               float* __restrict__ out,
               int M, int N, int K) {
#if USE_ASYNC_STAGING
    __shared__ unsigned short Asl[2][BM * LDT];
    __shared__ unsigned short Bsl[2][BM * LDT];
#else
    __shared__ unsigned short Asl[1][BM * LDT];
    __shared__ unsigned short Bsl[1][BM * LDT];
#endif

    const int m0   = blockIdx.y * BM;
    const int n0   = blockIdx.x * BN;
    const int tid  = threadIdx.x;
    const int wid  = tid >> 5;           // 8 waves
    const int lane = tid & 31;
    const int wm   = wid & 3;            // 4 waves along M (32 rows each)
    const int wn   = wid >> 2;           // 2 waves along N (64 cols each)

    f32x8 acc[2][4];
#pragma unroll
    for (int i = 0; i < 2; ++i)
#pragma unroll
        for (int j = 0; j < 4; ++j) acc[i][j] = {};

    // fragment lane mapping (ISA 7.12.2, 16-bit operands, wave32)
    const int fr  = lane & 15;            // row (A) / col (B) within fragment
    const int kbA = (lane >> 4) * 8;      // A: halves at K {kbA..kbA+7, 16+kbA..}
    const int kbB = (lane >> 4) * 8;      // B: halves at K {kbB..kbB+7, 16+kbB..}

    union Frag { bf16x16 v; bf16x8 h[2]; };

    // staging geometry: 512 chunks of 8 bf16 per 128x32 slab, 2 chunks/thread
    const int srow0 = (tid + 0)   >> 2, sseg0 = ((tid + 0)   & 3) * 8;
    const int srow1 = (tid + 256) >> 2, sseg1 = ((tid + 256) & 3) * 8;

#if USE_ASYNC_STAGING
    const int NT = K / BK;
    // prologue: stage slab 0 into buffer 0
    async_copy_b128(&A[(size_t)(m0 + srow0) * K + sseg0], &Asl[0][srow0 * LDT + sseg0]);
    async_copy_b128(&Bt[(size_t)(n0 + srow0) * K + sseg0], &Bsl[0][srow0 * LDT + sseg0]);
    async_copy_b128(&A[(size_t)(m0 + srow1) * K + sseg1], &Asl[0][srow1 * LDT + sseg1]);
    async_copy_b128(&Bt[(size_t)(n0 + srow1) * K + sseg1], &Bsl[0][srow1 * LDT + sseg1]);

    for (int kt = 0; kt < NT; ++kt) {
        const int cur = kt & 1;
        // all of this wave's async ops done; barrier makes tile visible to all
        asm volatile("s_wait_asynccnt 0x0" ::: "memory");
        __syncthreads();

        Frag af[2], bf[4];
#pragma unroll
        for (int i = 0; i < 2; ++i) {
            int row = wm * 32 + i * 16 + fr;
            af[i].h[0] = *(const bf16x8*)&Asl[cur][row * LDT + kbA];
            af[i].h[1] = *(const bf16x8*)&Asl[cur][row * LDT + 16 + kbA];
        }
#pragma unroll
        for (int j = 0; j < 4; ++j) {
            int col = wn * 64 + j * 16 + fr;
            bf[j].h[0] = *(const bf16x8*)&Bsl[cur][col * LDT + kbB];
            bf[j].h[1] = *(const bf16x8*)&Bsl[cur][col * LDT + kbB + 16];
        }

        // stage next slab into the other buffer; overlaps with the WMMAs below.
        // (writes to buf cur^1 are separated from its last reads by the barrier)
        if (kt + 1 < NT) {
            const int k0 = (kt + 1) * BK;
            const int nb = cur ^ 1;
            async_copy_b128(&A[(size_t)(m0 + srow0) * K + k0 + sseg0], &Asl[nb][srow0 * LDT + sseg0]);
            async_copy_b128(&Bt[(size_t)(n0 + srow0) * K + k0 + sseg0], &Bsl[nb][srow0 * LDT + sseg0]);
            async_copy_b128(&A[(size_t)(m0 + srow1) * K + k0 + sseg1], &Asl[nb][srow1 * LDT + sseg1]);
            async_copy_b128(&Bt[(size_t)(n0 + srow1) * K + k0 + sseg1], &Bsl[nb][srow1 * LDT + sseg1]);
        }

#pragma unroll
        for (int i = 0; i < 2; ++i)
#pragma unroll
            for (int j = 0; j < 4; ++j)
                acc[i][j] = __builtin_amdgcn_wmma_f32_16x16x32_bf16(
                    false, af[i].v, false, bf[j].v,
                    (short)0, acc[i][j], false, false);
    }
#else
    for (int k0 = 0; k0 < K; k0 += BK) {
        *(bf16x8*)&Asl[0][srow0 * LDT + sseg0] =
            *(const bf16x8*)&A[(size_t)(m0 + srow0) * K + k0 + sseg0];
        *(bf16x8*)&Bsl[0][srow0 * LDT + sseg0] =
            *(const bf16x8*)&Bt[(size_t)(n0 + srow0) * K + k0 + sseg0];
        *(bf16x8*)&Asl[0][srow1 * LDT + sseg1] =
            *(const bf16x8*)&A[(size_t)(m0 + srow1) * K + k0 + sseg1];
        *(bf16x8*)&Bsl[0][srow1 * LDT + sseg1] =
            *(const bf16x8*)&Bt[(size_t)(n0 + srow1) * K + k0 + sseg1];
        __syncthreads();

        Frag af[2], bf[4];
#pragma unroll
        for (int i = 0; i < 2; ++i) {
            int row = wm * 32 + i * 16 + fr;
            af[i].h[0] = *(const bf16x8*)&Asl[0][row * LDT + kbA];
            af[i].h[1] = *(const bf16x8*)&Asl[0][row * LDT + 16 + kbA];
        }
#pragma unroll
        for (int j = 0; j < 4; ++j) {
            int col = wn * 64 + j * 16 + fr;
            bf[j].h[0] = *(const bf16x8*)&Bsl[0][col * LDT + kbB];
            bf[j].h[1] = *(const bf16x8*)&Bsl[0][col * LDT + kbB + 16];
        }

#pragma unroll
        for (int i = 0; i < 2; ++i)
#pragma unroll
            for (int j = 0; j < 4; ++j)
                acc[i][j] = __builtin_amdgcn_wmma_f32_16x16x32_bf16(
                    false, af[i].v, false, bf[j].v,
                    (short)0, acc[i][j], false, false);
        __syncthreads();
    }
#endif

    // ---- epilogue: bias + relu, C/D layout (8 VGPR x 32 lanes) ----
    const int nsub = lane & 15;
    const int msub = (lane >> 4) * 8;
#pragma unroll
    for (int i = 0; i < 2; ++i) {
#pragma unroll
        for (int j = 0; j < 4; ++j) {
            int col = n0 + wn * 64 + j * 16 + nsub;
            float bv = bias[col];
#pragma unroll
            for (int r = 0; r < 8; ++r) {
                int row = m0 + wm * 32 + i * 16 + msub + r;
                float v = acc[i][j][r] + bv;
                v = v > 0.0f ? v : 0.0f;
                out[(size_t)row * N + col] = v;
            }
        }
    }
}

// ---------------- final row L2-normalize ----------------
__global__ void normalize_rows(const float* __restrict__ h,
                               float* __restrict__ out, int C) {
    __shared__ float red[256];
    int nid = blockIdx.x;
    const float* row = h + (size_t)nid * C;
    float ssum = 0.0f;
    for (int c = threadIdx.x; c < C; c += 256) {
        float v = row[c];
        ssum = fmaf(v, v, ssum);
    }
    red[threadIdx.x] = ssum;
    __syncthreads();
    for (int s = 128; s > 0; s >>= 1) {
        if (threadIdx.x < s) red[threadIdx.x] += red[threadIdx.x + s];
        __syncthreads();
    }
    float inv = 1.0f / fmaxf(sqrtf(red[0]), 1e-12f);
    for (int c = threadIdx.x; c < 256 * ((C + 255) / 256); c += 256)
        if (c < C) out[(size_t)nid * C + c] = row[c] * inv;
}

// ---------------- host orchestration ----------------
extern "C" void kernel_launch(void* const* d_in, const int* in_sizes, int n_in,
                              void* d_out, int out_size, void* d_ws, size_t ws_size,
                              hipStream_t stream) {
    const float* x     = (const float*)d_in[0];   // [NN, FIN]
    const float* pos   = (const float*)d_in[1];   // [NN, 2]
    const int*   ei    = (const int*)  d_in[2];   // [2, NE]
    const float* W_in1 = (const float*)d_in[3];
    const float* b_in1 = (const float*)d_in[4];
    const float* W_o1  = (const float*)d_in[5];
    const float* b_o1  = (const float*)d_in[6];
    const float* W_in2 = (const float*)d_in[7];
    const float* b_in2 = (const float*)d_in[8];
    const float* W_o2  = (const float*)d_in[9];
    const float* b_o2  = (const float*)d_in[10];
    const float* W_in3 = (const float*)d_in[11];
    const float* b_in3 = (const float*)d_in[12];
    const float* W_o3  = (const float*)d_in[13];
    const float* b_o3  = (const float*)d_in[14];
    const int* src = ei;
    const int* dst = ei + NE;

    // workspace carve-out (256B aligned)
    char* ws = (char*)d_ws;
    size_t off = 0;
    auto carve = [&](size_t bytes) -> void* {
        void* p = ws + off;
        off = (off + bytes + 255) & ~(size_t)255;
        return p;
    };
    float*          buf0 = (float*)         carve((size_t)MPAD * 2048 * sizeof(float));
    float*          buf1 = (float*)         carve((size_t)MPAD * 2048 * sizeof(float));
    unsigned short* Abf  = (unsigned short*)carve((size_t)MPAD * 2048 * sizeof(unsigned short));
    unsigned short* Wt   = (unsigned short*)carve((size_t)2048 * 2048 * sizeof(unsigned short));

    auto run_layer = [&](const float* xin, int ldx,
                         const float* Wi, const float* bi,
                         const float* Wo, const float* bo,
                         float* agg, float* hout, int Cin, int Cout) {
        size_t na = (size_t)MPAD * Cin;
        zero_f32<<<2048, 256, 0, stream>>>(agg, na);
        edge_scatter<<<NE, 256, 0, stream>>>(xin, ldx, pos, src, dst,
                                             Wi, bi, agg, Cin, Cin);
        cvt_f32_to_bf16<<<2048, 256, 0, stream>>>(agg, Abf, na);
        cvt_transpose_w<<<2048, 256, 0, stream>>>(Wo, Wt, Cin, Cout);
        dim3 grid(Cout / BN, MPAD / BM);
        gemm_bf16_relu<<<grid, 256, 0, stream>>>(Abf, Wt, bo, hout,
                                                 MPAD, Cout, Cin);
    };

    // layer 1: x [NN,128] -> h1 [MPAD,2048] (agg1 lives in buf1, overwritten by h1)
    run_layer(x,    FIN,  W_in1, b_in1, W_o1, b_o1, buf1, buf1, FIN, H1);
    // layer 2: h1 -> h2 [MPAD,2048]
    run_layer(buf1, H1,   W_in2, b_in2, W_o2, b_o2, buf0, buf0, H1,  H2);
    // layer 3: h2 -> h3 [MPAD,1024]
    run_layer(buf0, H2,   W_in3, b_in3, W_o3, b_o3, buf1, buf1, H2,  H3);

    // final L2 normalize of first NN rows into d_out [NN,1024]
    normalize_rows<<<NN, 256, 0, stream>>>(buf1, (float*)d_out, H3);
}
